// StanceDetectionModel13_67662914781724
// MI455X (gfx1250) — matrix operand order repaired and checked by reading, not compiled, not run on hardware
//
#include <hip/hip_runtime.h>
#include <hip/hip_bf16.h>

// ---------------------------------------------------------------------------
// StanceDetection pipeline for MI455X (gfx1250, wave32, WMMA bf16).
//
// Kernels:
//   1. embed_kernel          : E_word[tok] + E_pos[t] -> bf16 rows
//   2. transpose_bf16_kernel : W[K,N] f32 -> W^T[N,K] bf16 (WMMA B layout)
//   3. gru_kernel            : fused minGRU recurrence, WMMA 16x16x32 bf16,
//                              hidden state double-buffered in LDS,
//                              split K-phases (global x-part / LDS h-part),
//                              A-fragment hoisted across 6 N-fragments
//   4. gemm_kernel           : generic WMMA GEMM w/ bias + activation epilogue
//   5. attn_score_kernel     : fused tanh-GEMM + dot(w2) via LDS ds_add_f32
//   6. softmax_context_kernel: softmax over T + attention-weighted pooling
//   7. logits_kernel         : tiny [64,768]x[768,3] head
// ---------------------------------------------------------------------------

#define BB   64
#define STT  512
#define SKK  32
#define DD   512
#define HH   768

typedef __attribute__((ext_vector_type(16))) __bf16 bf16x16;
typedef __attribute__((ext_vector_type(8)))  float  f32x8;

union FragCast { bf16x16 v; uint4 q[2]; };

// A-matrix fragment (16x32 bf16, MxK): lane holds row m=lane&15,
// K chunks [k0+off, +8) and [k0+16+off, +8), off = (lane>=16)?8:0.
__device__ __forceinline__ bf16x16 load_frag_a(const __bf16* base, long ld,
                                               int row0, int k0) {
  const int lane = threadIdx.x & 31;
  const int r    = row0 + (lane & 15);
  const int off  = (lane & 16) ? 8 : 0;
  const __bf16* p = base + (long)r * ld + k0 + off;
  FragCast f;
  f.q[0] = *(const uint4*)(p);
  f.q[1] = *(const uint4*)(p + 16);
  return f.v;
}

// B-matrix fragment (32x16 bf16, KxN) from pre-transposed weights WT[N][K]:
// lane holds col n=lane&15, K chunks [k0+off, +8) and [k0+16+off, +8), off = (lane>=16)?8:0.
__device__ __forceinline__ bf16x16 load_frag_b(const __bf16* base, long ld,
                                               int col0, int k0) {
  const int lane = threadIdx.x & 31;
  const int n    = col0 + (lane & 15);
  const int off  = (lane & 16) ? 8 : 0;
  const __bf16* p = base + (long)n * ld + k0 + off;
  FragCast f;
  f.q[0] = *(const uint4*)(p);
  f.q[1] = *(const uint4*)(p + 16);
  return f.v;
}

__device__ __forceinline__ f32x8 wmma_bf16(bf16x16 a, bf16x16 b, f32x8 c) {
  return __builtin_amdgcn_wmma_f32_16x16x32_bf16(
      /*neg_a=*/false, a, /*neg_b=*/false, b,
      /*c_mod=*/(short)0, c, /*reuse_a=*/false, /*reuse_b=*/false);
}

__device__ __forceinline__ float sigmoidf_dev(float x) {
  return 1.0f / (1.0f + __expf(-x));
}

// ---------------------------------------------------------------------------
// 1. Embedding: out[r][d] = E_word[tok[r]][d] + E_pos[r % S][d]  (bf16)
// ---------------------------------------------------------------------------
__global__ void embed_kernel(const int* __restrict__ tok,
                             const float* __restrict__ Ew,
                             const float* __restrict__ Ep,
                             __bf16* __restrict__ out, int S) {
  const int r = blockIdx.x;
  const int t = r % S;                 // S <= MAXP, no clip needed
  const int token = tok[r];
  const float* w = Ew + (size_t)token * DD;
  const float* p = Ep + (size_t)t * DD;
  __bf16* o = out + (size_t)r * DD;
  for (int d = threadIdx.x; d < DD; d += blockDim.x)
    o[d] = (__bf16)(w[d] + p[d]);
}

// ---------------------------------------------------------------------------
// 2. Weight transpose + downconvert: WT[n][k] = (bf16) W[k + koff][n]
// ---------------------------------------------------------------------------
__global__ void transpose_bf16_kernel(const float* __restrict__ W,
                                      __bf16* __restrict__ WT,
                                      int K, int N, int koff) {
  const int idx = blockIdx.x * blockDim.x + threadIdx.x;
  if (idx >= K * N) return;
  const int n = idx / K;
  const int k = idx % K;
  WT[idx] = (__bf16)W[(size_t)(k + koff) * N + n];
}

// ---------------------------------------------------------------------------
// 3. Fused minGRU. Block = 256 thr (8 waves), owns 16 batch rows.
//    Per step, per wave: 6 column fragments (N=96 slice), 6x2 persistent
//    f32 accumulators; phase 1 streams A from global embeddings (K=512),
//    phase 2 streams A from the LDS hidden state (K=768). One A-load feeds
//    12 WMMAs. Hidden state double-buffered in LDS as bf16 (2 x 24KB).
// ---------------------------------------------------------------------------
__global__ __launch_bounds__(256) void gru_kernel(
    const __bf16* __restrict__ emb, int S,          // [B][S][DD] bf16
    const __bf16* __restrict__ WzT,                 // [HH][DD+HH] bf16
    const __bf16* __restrict__ WhT,                 // [HH][DD+HH] bf16
    const float* __restrict__ bz, const float* __restrict__ bh,
    float* __restrict__ out_seq,                    // [B][S][HH] f32 or null
    __bf16* __restrict__ out_seq_bf,                // [B][S][HH] bf16 or null
    float* __restrict__ h_final,                    // [B][HH] f32 or null
    __bf16* __restrict__ h_final_bf, int hfb_ld) {
  __shared__ __bf16 hbuf[2][16 * HH];

  const int tid  = threadIdx.x;
  const int lane = tid & 31;
  const int wave = tid >> 5;
  const int b0   = blockIdx.x * 16;

  for (int i = tid; i < 16 * HH; i += 256) {
    hbuf[0][i] = (__bf16)0.0f;
    hbuf[1][i] = (__bf16)0.0f;
  }
  __syncthreads();

  const int nbase = wave * 96;         // this wave's N slice
  const int n_col = (lane & 15);
  const int mbase = (lane & 16) ? 8 : 0;
  const f32x8 zero = {};

  for (int t = 0; t < S; ++t) {
    const __bf16* cur = &hbuf[t & 1][0];
    __bf16*       nxt = &hbuf[(t & 1) ^ 1][0];
    const __bf16* aGlob = emb + ((size_t)b0 * S + t) * DD;
    const long    ldaG  = (long)S * DD;

    f32x8 accz[6], acch[6];
#pragma unroll
    for (int nf = 0; nf < 6; ++nf) { accz[nf] = zero; acch[nf] = zero; }

    // ---- phase 1: x-part (A from global embeddings, K = 0..511) ----
    for (int kk = 0; kk < DD; kk += 32) {
      bf16x16 a = load_frag_a(aGlob, ldaG, 0, kk);
#pragma unroll
      for (int nf = 0; nf < 6; ++nf) {
        const int n0 = nbase + nf * 16;
        accz[nf] = wmma_bf16(a, load_frag_b(WzT, DD + HH, n0, kk), accz[nf]);
        acch[nf] = wmma_bf16(a, load_frag_b(WhT, DD + HH, n0, kk), acch[nf]);
      }
    }
    // ---- phase 2: h-part (A from LDS hidden state, K = 512..1279) ----
    for (int kk = 0; kk < HH; kk += 32) {
      bf16x16 a = load_frag_a(cur, HH, 0, kk);
#pragma unroll
      for (int nf = 0; nf < 6; ++nf) {
        const int n0 = nbase + nf * 16;
        accz[nf] = wmma_bf16(a, load_frag_b(WzT, DD + HH, n0, DD + kk), accz[nf]);
        acch[nf] = wmma_bf16(a, load_frag_b(WhT, DD + HH, n0, DD + kk), acch[nf]);
      }
    }

    // ---- epilogue: gate, update hidden state, emit outputs ----
#pragma unroll
    for (int nf = 0; nf < 6; ++nf) {
      const int   n  = nbase + nf * 16 + n_col;
      const float vz = bz[n];
      const float vh = bh[n];
#pragma unroll
      for (int i = 0; i < 8; ++i) {
        const int m = i + mbase;
        const float z    = sigmoidf_dev(accz[nf][i] + vz);
        const float htld = tanhf(acch[nf][i] + vh);
        const float hold = (float)cur[m * HH + n];
        const float hnew = (1.0f - z) * hold + z * htld;
        nxt[m * HH + n] = (__bf16)hnew;
        if (out_seq) {
          const size_t o = ((size_t)(b0 + m) * S + t) * HH + n;
          out_seq[o]    = hnew;
          out_seq_bf[o] = (__bf16)hnew;
        }
      }
    }
    __syncthreads();   // all reads of `cur` done; `nxt` fully written
  }

  if (h_final) {
    const __bf16* fin = &hbuf[S & 1][0];
    for (int i = tid; i < 16 * HH; i += 256) {
      const int m = i / HH, n = i % HH;
      const float v = (float)fin[i];
      h_final[(size_t)(b0 + m) * HH + n] = v;
      h_final_bf[(size_t)(b0 + m) * hfb_ld + n] = (__bf16)v;
    }
  }
}

// ---------------------------------------------------------------------------
// 4. Generic WMMA GEMM: C[M,N] = act(A[M,K](bf16) @ B + bias)
//    B supplied transposed: BT[N][K]. One 16x16 tile per wave.
// ---------------------------------------------------------------------------
__global__ __launch_bounds__(256) void gemm_kernel(
    const __bf16* __restrict__ A, long lda,
    const __bf16* __restrict__ BT, long ldb,
    const float* __restrict__ bias,
    float* __restrict__ C, int M, int N, int K, int act) {
  const int lane = threadIdx.x & 31;
  const int wave = threadIdx.x >> 5;
  const int tn   = N >> 4;
  const int tile = blockIdx.x * 8 + wave;
  if (tile >= (M >> 4) * tn) return;   // wave-uniform exit, EXEC stays full
  const int mt = tile / tn;
  const int nt = tile % tn;

  f32x8 acc = {};
  for (int kk = 0; kk < K; kk += 32) {
    bf16x16 a = load_frag_a(A, lda, mt * 16, kk);
    bf16x16 b = load_frag_b(BT, ldb, nt * 16, kk);
    if (kk + 32 < K)
      __builtin_prefetch(BT + (long)(nt * 16) * ldb + kk + 32, 0, 1);
    acc = wmma_bf16(a, b, acc);
  }
  const int   n  = nt * 16 + (lane & 15);
  const float bv = bias ? bias[n] : 0.0f;
  const int   mbase = (lane & 16) ? 8 : 0;
#pragma unroll
  for (int i = 0; i < 8; ++i) {
    const int m = mt * 16 + i + mbase;
    float v = acc[i] + bv;
    if (act == 1)      v = tanhf(v);
    else if (act == 2) v = fmaxf(v, 0.0f);
    C[(size_t)m * N + n] = v;
  }
}

// ---------------------------------------------------------------------------
// 5. Fused attention scores: s[row] = tanh(A[row]@W1a + u[b] + b1) . w2 + b2
//    Block owns 64 rows (one batch b), all N=768, K=768; A-fragment hoisted
//    across the wave's 6 column fragments; dot with w2 reduced through LDS
//    float atomics (ds_add_f32).
// ---------------------------------------------------------------------------
__global__ __launch_bounds__(256) void attn_score_kernel(
    const __bf16* __restrict__ Abf,      // [B*ST][HH] bf16 (text_out)
    const __bf16* __restrict__ W1aT,     // [HH][HH]
    const float* __restrict__ u,         // [B][HH]  (topic_h @ W1b)
    const float* __restrict__ b1,
    const float* __restrict__ w2,
    const float* __restrict__ b2v,
    float* __restrict__ scores) {        // [B*ST]
  __shared__ float sacc[64];
  const int tid  = threadIdx.x;
  const int lane = tid & 31;
  const int wave = tid >> 5;
  const int row0 = blockIdx.x * 64;
  const int b    = row0 / STT;

  if (tid < 64) sacc[tid] = 0.0f;
  __syncthreads();

  const int nbase = wave * 96;
  const int mbase = (lane & 16) ? 8 : 0;
  const f32x8 zero = {};

  for (int mf = 0; mf < 4; ++mf) {
    f32x8 acc[6];
#pragma unroll
    for (int j = 0; j < 6; ++j) acc[j] = zero;

    for (int kk = 0; kk < HH; kk += 32) {
      bf16x16 a = load_frag_a(Abf, HH, row0 + mf * 16, kk);
#pragma unroll
      for (int j = 0; j < 6; ++j)
        acc[j] = wmma_bf16(a, load_frag_b(W1aT, HH, nbase + j * 16, kk), acc[j]);
    }

#pragma unroll
    for (int j = 0; j < 6; ++j) {
      const int   n   = nbase + j * 16 + (lane & 15);
      const float ub  = u[(size_t)b * HH + n] + b1[n];
      const float w2n = w2[n];
#pragma unroll
      for (int i = 0; i < 8; ++i) {
        const int ml = mf * 16 + i + mbase;
        const float v = tanhf(acc[j][i] + ub);
        atomicAdd(&sacc[ml], v * w2n);     // ds_add_f32
      }
    }
  }
  __syncthreads();
  if (tid < 64) scores[row0 + tid] = sacc[tid] + b2v[0];
}

// ---------------------------------------------------------------------------
// 6. Softmax over T + context pooling. One block per batch row.
// ---------------------------------------------------------------------------
__global__ __launch_bounds__(256) void softmax_context_kernel(
    const float* __restrict__ scores,    // [B][ST]
    const float* __restrict__ text_out,  // [B][ST][HH] f32
    float* __restrict__ attn_out,        // [B][ST] (into d_out)
    float* __restrict__ context,         // [B][HH]
    __bf16* __restrict__ feats) {        // [B][2*HH], writes first HH
  __shared__ float sv[STT];
  __shared__ float red[256];
  const int b   = blockIdx.x;
  const int tid = threadIdx.x;

  float lm = -3.4e38f;
  for (int t = tid; t < STT; t += 256) {
    const float s = scores[b * STT + t];
    sv[t] = s;
    lm = fmaxf(lm, s);
  }
  red[tid] = lm;
  __syncthreads();
  for (int s = 128; s > 0; s >>= 1) {
    if (tid < s) red[tid] = fmaxf(red[tid], red[tid + s]);
    __syncthreads();
  }
  const float mx = red[0];
  __syncthreads();

  float ls = 0.0f;
  for (int t = tid; t < STT; t += 256) {
    const float e = __expf(sv[t] - mx);
    sv[t] = e;
    ls += e;
  }
  red[tid] = ls;
  __syncthreads();
  for (int s = 128; s > 0; s >>= 1) {
    if (tid < s) red[tid] += red[tid + s];
    __syncthreads();
  }
  const float inv = 1.0f / red[0];
  __syncthreads();

  for (int t = tid; t < STT; t += 256) {
    const float a = sv[t] * inv;
    sv[t] = a;
    attn_out[b * STT + t] = a;
  }
  __syncthreads();

  for (int h = tid; h < HH; h += 256) {
    float acc = 0.0f;
    const float* src = text_out + (size_t)b * STT * HH + h;
    for (int t = 0; t < STT; ++t) acc += sv[t] * src[(size_t)t * HH];
    context[(size_t)b * HH + h] = acc;
    feats[(size_t)b * (2 * HH) + h] = (__bf16)acc;
  }
}

// ---------------------------------------------------------------------------
// 7. Logits head: [64,768] x [768,3] + bias
// ---------------------------------------------------------------------------
__global__ void logits_kernel(const float* __restrict__ x1,
                              const float* __restrict__ Wf2,
                              const float* __restrict__ bf2,
                              float* __restrict__ out) {
  const int tid = blockIdx.x * blockDim.x + threadIdx.x;
  if (tid >= BB * 3) return;
  const int bb = tid / 3, c = tid % 3;
  float acc = bf2[c];
  const float* xr = x1 + (size_t)bb * HH;
  for (int h = 0; h < HH; ++h) acc += xr[h] * Wf2[h * 3 + c];
  out[tid] = acc;
}

// ---------------------------------------------------------------------------
extern "C" void kernel_launch(void* const* d_in, const int* in_sizes, int n_in,
                              void* d_out, int out_size, void* d_ws, size_t ws_size,
                              hipStream_t stream) {
  const int*   text   = (const int*)d_in[0];
  const int*   topic  = (const int*)d_in[1];
  const float* E_word = (const float*)d_in[2];
  const float* E_pos  = (const float*)d_in[3];
  const float* Wz_t = (const float*)d_in[4],  *bz_t = (const float*)d_in[5];
  const float* Wh_t = (const float*)d_in[6],  *bh_t = (const float*)d_in[7];
  const float* Wz_k = (const float*)d_in[8],  *bz_k = (const float*)d_in[9];
  const float* Wh_k = (const float*)d_in[10], *bh_k = (const float*)d_in[11];
  const float* W1   = (const float*)d_in[12], *b1   = (const float*)d_in[13];
  const float* w2   = (const float*)d_in[14], *b2   = (const float*)d_in[15];
  const float* Wf1  = (const float*)d_in[16], *bf1  = (const float*)d_in[17];
  const float* Wf2  = (const float*)d_in[18], *bf2  = (const float*)d_in[19];
  float* out = (float*)d_out;                 // [192 logits][32768 attn]

  // ---- workspace layout ----
  char* ws = (char*)d_ws;
  size_t o = 0;
  auto alloc = [&](size_t bytes) {
    size_t r = o;
    o = (o + bytes + 255) & ~(size_t)255;
    return r;
  };
  __bf16* text_emb    = (__bf16*)(ws + alloc((size_t)BB * STT * DD * 2));
  __bf16* topic_emb   = (__bf16*)(ws + alloc((size_t)BB * SKK * DD * 2));
  __bf16* WzTt        = (__bf16*)(ws + alloc((size_t)HH * (DD + HH) * 2));
  __bf16* WhTt        = (__bf16*)(ws + alloc((size_t)HH * (DD + HH) * 2));
  __bf16* WzTk        = (__bf16*)(ws + alloc((size_t)HH * (DD + HH) * 2));
  __bf16* WhTk        = (__bf16*)(ws + alloc((size_t)HH * (DD + HH) * 2));
  __bf16* W1aT        = (__bf16*)(ws + alloc((size_t)HH * HH * 2));
  __bf16* W1bT        = (__bf16*)(ws + alloc((size_t)HH * HH * 2));
  __bf16* Wf1T        = (__bf16*)(ws + alloc((size_t)HH * 2 * HH * 2));
  float*  text_out    = (float*)(ws + alloc((size_t)BB * STT * HH * 4));
  __bf16* text_out_bf = (__bf16*)(ws + alloc((size_t)BB * STT * HH * 2));
  float*  topic_h     = (float*)(ws + alloc((size_t)BB * HH * 4));
  __bf16* feats       = (__bf16*)(ws + alloc((size_t)BB * 2 * HH * 2));
  float*  u           = (float*)(ws + alloc((size_t)BB * HH * 4));
  float*  scores      = (float*)(ws + alloc((size_t)BB * STT * 4));
  float*  context     = (float*)(ws + alloc((size_t)BB * HH * 4));
  float*  x1          = (float*)(ws + alloc((size_t)BB * HH * 4));

  // ---- 1. embeddings (bf16) ----
  embed_kernel<<<BB * STT, 128, 0, stream>>>(text, E_word, E_pos, text_emb, STT);
  embed_kernel<<<BB * SKK, 128, 0, stream>>>(topic, E_word, E_pos, topic_emb, SKK);

  // ---- 2. weight transposes (f32 -> bf16, [N][K]) ----
  const int grK = ((DD + HH) * HH + 255) / 256;   // 1280x768
  transpose_bf16_kernel<<<grK, 256, 0, stream>>>(Wz_t, WzTt, DD + HH, HH, 0);
  transpose_bf16_kernel<<<grK, 256, 0, stream>>>(Wh_t, WhTt, DD + HH, HH, 0);
  transpose_bf16_kernel<<<grK, 256, 0, stream>>>(Wz_k, WzTk, DD + HH, HH, 0);
  transpose_bf16_kernel<<<grK, 256, 0, stream>>>(Wh_k, WhTk, DD + HH, HH, 0);
  const int grH = (HH * HH + 255) / 256;          // 768x768
  transpose_bf16_kernel<<<grH, 256, 0, stream>>>(W1, W1aT, HH, HH, 0);
  transpose_bf16_kernel<<<grH, 256, 0, stream>>>(W1, W1bT, HH, HH, HH);
  const int grF = (2 * HH * HH + 255) / 256;      // 1536x768
  transpose_bf16_kernel<<<grF, 256, 0, stream>>>(Wf1, Wf1T, 2 * HH, HH, 0);

  // ---- 3. minGRU recurrences (4 blocks x 16 batch rows each) ----
  gru_kernel<<<BB / 16, 256, 0, stream>>>(
      text_emb, STT, WzTt, WhTt, bz_t, bh_t,
      text_out, text_out_bf, nullptr, nullptr, 0);
  gru_kernel<<<BB / 16, 256, 0, stream>>>(
      topic_emb, SKK, WzTk, WhTk, bz_k, bh_k,
      nullptr, nullptr, topic_h, feats + HH, 2 * HH);

  // ---- 4. u = topic_h @ W1b   (A = bf16 topic_h inside feats, lda=1536) ----
  {
    const int tiles = (BB / 16) * (HH / 16);
    gemm_kernel<<<(tiles + 7) / 8, 256, 0, stream>>>(
        feats + HH, 2 * HH, W1bT, HH, nullptr, u, BB, HH, HH, 0);
  }

  // ---- 5. attention scores (fused tanh-GEMM + dot w2) ----
  attn_score_kernel<<<BB * STT / 64, 256, 0, stream>>>(
      text_out_bf, W1aT, u, b1, w2, b2, scores);

  // ---- 6. softmax + context; attn written straight into d_out ----
  softmax_context_kernel<<<BB, 256, 0, stream>>>(
      scores, text_out, out + BB * 3, context, feats);

  // ---- 7. classifier: x1 = relu(feats @ Wf1 + bf1); logits ----
  {
    const int tiles = (BB / 16) * (HH / 16);
    gemm_kernel<<<(tiles + 7) / 8, 256, 0, stream>>>(
        feats, 2 * HH, Wf1T, 2 * HH, bf1, x1, BB, HH, 2 * HH, 2);
  }
  logits_kernel<<<1, 256, 0, stream>>>(x1, Wf2, bf2, out);
}
